// MultiKANProjection_23321672417779
// MI455X (gfx1250) — matrix operand order repaired and checked by reading, not compiled, not run on hardware
//
#include <hip/hip_runtime.h>

// ---------------------------------------------------------------------------
// MultiKAN projection for MI455X (gfx1250).
// Restructured as: activations -> bf16 pack -> single bf16 WMMA GEMM
//   C[8192 x 3072] = A[8192 x 6912] * W^T,  A = [silu(x) | bsplines],
//   W rows = [base_w | spline_w.flat], heads 768/768/1536 concat on N.
// Compute-bound (AI ~1650 FLOP/B) -> v_wmma_f32_16x16x32_bf16 path.
// This version double-buffers the LDS tiles: one barrier per K-step and
// global loads for step k+1 overlap the WMMA stream of step k.
// ---------------------------------------------------------------------------

typedef __bf16 bf16_t;
typedef __attribute__((ext_vector_type(8)))  __bf16 v8bf;
typedef __attribute__((ext_vector_type(16))) __bf16 v16bf;
typedef __attribute__((ext_vector_type(8)))  float  v8f;

// Problem constants (from the reference)
constexpr int IN    = 768;
constexpr int GK    = 8;            // G + K spline bases per input
constexpr int NKNOT = 12;           // G + 2K + 1 knots
constexpr int NROWS = 8192;         // B * S
constexpr int KTOT  = IN + IN * GK; // 6912 packed reduction dim
constexpr int O0 = 768, O1 = 768, O2 = 1536;
constexpr int OTOT  = O0 + O1 + O2; // 3072

// GEMM tiling
constexpr int BM = 128, BN = 128, BK = 32;
constexpr int LDT = 40;             // padded LDS row (bf16 elems) to spread banks
constexpr int KT  = KTOT / BK;      // 216 K-steps

// ---------------------------------------------------------------------------
// Kernel 1: per (row, input-feature) compute SiLU base + 8 cubic B-spline
// bases (Cox-de Boor, fp32), pack to bf16 activation matrix Ap[N x 6912].
// Row layout: [ silu(x_0..767) | bs(i=0,k=0..7), bs(i=1,...), ... ]
// ---------------------------------------------------------------------------
__global__ void kan_activations(const float* __restrict__ x,
                                const float* __restrict__ grid,
                                bf16_t* __restrict__ Ap) {
  int tid = blockIdx.x * blockDim.x + threadIdx.x;
  if (tid >= NROWS * IN) return;
  int n = tid / IN;
  int i = tid - n * IN;

  float xv = x[tid];
  const float* g = grid + i * NKNOT;
  float gk[NKNOT];
#pragma unroll
  for (int t = 0; t < NKNOT; ++t) gk[t] = g[t];

  float b[NKNOT - 1];
#pragma unroll
  for (int t = 0; t < NKNOT - 1; ++t)
    b[t] = (xv >= gk[t] && xv < gk[t + 1]) ? 1.0f : 0.0f;

#pragma unroll
  for (int p = 1; p <= 3; ++p) {
#pragma unroll
    for (int t = 0; t < NKNOT - 2; ++t) {          // p is constant after unroll
      if (t < NKNOT - 1 - p) {
        float left  = (xv - gk[t])         / (gk[t + p]     - gk[t]);
        float right = (gk[t + p + 1] - xv) / (gk[t + p + 1] - gk[t + 1]);
        b[t] = left * b[t] + right * b[t + 1];
      }
    }
  }

  float s = xv / (1.0f + __expf(-xv)); // silu

  bf16_t* row = Ap + (size_t)n * KTOT;
  row[i] = (bf16_t)s;
  v8bf v;
#pragma unroll
  for (int k = 0; k < GK; ++k) v[k] = (bf16_t)b[k];
  *(v8bf*)(row + IN + i * GK) = v;     // 16B aligned: (768 + 8i)*2, row 13824B
}

// ---------------------------------------------------------------------------
// Kernel 2: pack the 3 heads' [base_w | spline_w] into bf16 Wp[3072 x 6912].
// ---------------------------------------------------------------------------
__global__ void kan_pack_weights(const float* __restrict__ bw0,
                                 const float* __restrict__ sw0,
                                 const float* __restrict__ bw1,
                                 const float* __restrict__ sw1,
                                 const float* __restrict__ bw2,
                                 const float* __restrict__ sw2,
                                 bf16_t* __restrict__ Wp) {
  long tid = (long)blockIdx.x * blockDim.x + threadIdx.x;
  if (tid >= (long)OTOT * KTOT) return;
  int o = (int)(tid / KTOT);
  int k = (int)(tid - (long)o * KTOT);

  const float* bw; const float* sw; int ol;
  if (o < O0)           { bw = bw0; sw = sw0; ol = o; }
  else if (o < O0 + O1) { bw = bw1; sw = sw1; ol = o - O0; }
  else                  { bw = bw2; sw = sw2; ol = o - O0 - O1; }

  float v = (k < IN) ? bw[(size_t)ol * IN + k]
                     : sw[(size_t)ol * (IN * GK) + (k - IN)];
  Wp[tid] = (bf16_t)v;
}

// ---------------------------------------------------------------------------
// Kernel 3: bf16 WMMA GEMM, double-buffered LDS. Block = 256 thr (8 wave32),
// tile 128x128x32. Waves 4(M) x 2(N); each wave: 32x64 region = 2x4 WMMA tiles.
// A frag (16x32 bf16): lane m<16 holds K{0..7,16..23}, lane m+16 K{8..15,24..31}.
// B frag (32x16 bf16): lane n<16 holds K{0..7,16..23}, lane n+16 K{8..15,24..31}.
// C/D: VGPR j -> row j (lanes 0-15) / row 8+j (lanes 16-31), col = lane&15.
// ---------------------------------------------------------------------------
__global__ __launch_bounds__(256)
void kan_gemm(const bf16_t* __restrict__ Ap,
              const bf16_t* __restrict__ Wp,
              float* __restrict__ out) {
  __shared__ bf16_t As[2][BM * LDT];   // 2 x 10240 B
  __shared__ bf16_t Bs[2][BN * LDT];   // 2 x 10240 B

  const int t    = threadIdx.x;
  const int lane = t & 31;
  const int wave = t >> 5;
  const int wm   = wave & 3;     // 0..3 -> M sub-blocks of 32
  const int wn   = wave >> 2;    // 0..1 -> N sub-blocks of 64
  const int l15  = lane & 15;
  const int lh   = lane >> 4;    // lane half

  const int m0 = blockIdx.y * BM;
  const int o0 = blockIdx.x * BN;

  // cooperative tile-load mapping: 64 rows x 32 cols per pass, 2 passes
  const int lr = t >> 2;          // 0..63
  const int lc = (t & 3) * 8;     // 0,8,16,24 (bf16 elems; 16B chunks)

  const bf16_t* aG = Ap + (size_t)(m0 + lr) * KTOT + lc;
  const bf16_t* bG = Wp + (size_t)(o0 + lr) * KTOT + lc;

  // per-wave LDS fragment offsets (element units, buffer-relative)
  const int aOff0 = (wm * 32 + 0 * 16 + l15) * LDT + lh * 8;
  const int aOff1 = (wm * 32 + 1 * 16 + l15) * LDT + lh * 8;
  const int bOffBase = (wn * 64 + l15) * LDT + lh * 8;

  v8f acc[2][4];
#pragma unroll
  for (int i = 0; i < 2; ++i)
#pragma unroll
    for (int j = 0; j < 4; ++j) { v8f z = {}; acc[i][j] = z; }

  // ---- prologue: stage tile 0 into buffer 0 ----
  {
    uint4 a0 = *(const uint4*)(aG);
    uint4 a1 = *(const uint4*)(aG + (size_t)64 * KTOT);
    uint4 b0 = *(const uint4*)(bG);
    uint4 b1 = *(const uint4*)(bG + (size_t)64 * KTOT);
    *(uint4*)(&As[0][lr * LDT + lc])        = a0;
    *(uint4*)(&As[0][(lr + 64) * LDT + lc]) = a1;
    *(uint4*)(&Bs[0][lr * LDT + lc])        = b0;
    *(uint4*)(&Bs[0][(lr + 64) * LDT + lc]) = b1;
  }
  __syncthreads();

  int p = 0;
  for (int kt = 0; kt < KT; ++kt) {
    const bool have_next = (kt + 1 < KT);

    // issue global loads for tile kt+1 early; they complete under the WMMAs
    uint4 a0n, a1n, b0n, b1n;
    if (have_next) {
      const int kb = (kt + 1) * BK;
      a0n = *(const uint4*)(aG + kb);
      a1n = *(const uint4*)(aG + (size_t)64 * KTOT + kb);
      b0n = *(const uint4*)(bG + kb);
      b1n = *(const uint4*)(bG + (size_t)64 * KTOT + kb);
    }
    if (kt + 2 < KT) {            // global_prefetch of the tile after that
      const int kb2 = (kt + 2) * BK;
      __builtin_prefetch(aG + kb2, 0, 1);
      __builtin_prefetch(bG + kb2, 0, 1);
    }

    // ---- compute on buffer p ----
    const bf16_t* Ab = As[p];
    const bf16_t* Bb = Bs[p];

    v16bf afr[2];
    {
      const bf16_t* pa = Ab + aOff0;
      v8bf lo = *(const v8bf*)pa;
      v8bf hi = *(const v8bf*)(pa + 16);
      afr[0] = __builtin_shufflevector(lo, hi, 0, 1, 2, 3, 4, 5, 6, 7,
                                       8, 9, 10, 11, 12, 13, 14, 15);
      pa = Ab + aOff1;
      lo = *(const v8bf*)pa;
      hi = *(const v8bf*)(pa + 16);
      afr[1] = __builtin_shufflevector(lo, hi, 0, 1, 2, 3, 4, 5, 6, 7,
                                       8, 9, 10, 11, 12, 13, 14, 15);
    }
    v16bf bfr[4];
#pragma unroll
    for (int tn = 0; tn < 4; ++tn) {
      const bf16_t* pb = Bb + bOffBase + tn * 16 * LDT;
      v8bf lo = *(const v8bf*)pb;
      v8bf hi = *(const v8bf*)(pb + 16);
      bfr[tn] = __builtin_shufflevector(lo, hi, 0, 1, 2, 3, 4, 5, 6, 7,
                                        8, 9, 10, 11, 12, 13, 14, 15);
    }

#pragma unroll
    for (int tm = 0; tm < 2; ++tm)
#pragma unroll
      for (int tn = 0; tn < 4; ++tn)
        acc[tm][tn] = __builtin_amdgcn_wmma_f32_16x16x32_bf16(
            false, afr[tm], false, bfr[tn], (short)0, acc[tm][tn],
            false, false);

    // ---- stage tile kt+1 into the other buffer; single barrier ----
    if (have_next) {
      const int q = p ^ 1;
      *(uint4*)(&As[q][lr * LDT + lc])        = a0n;
      *(uint4*)(&As[q][(lr + 64) * LDT + lc]) = a1n;
      *(uint4*)(&Bs[q][lr * LDT + lc])        = b0n;
      *(uint4*)(&Bs[q][(lr + 64) * LDT + lc]) = b1n;
      __syncthreads();
      p = q;
    }
  }

  // Epilogue: this N tile lies entirely inside one head (768 % 128 == 0)
  size_t outBase; int width; int oc0;
  if (o0 < O0)           { outBase = 0;                         width = O0; oc0 = o0; }
  else if (o0 < O0 + O1) { outBase = (size_t)NROWS * O0;        width = O1; oc0 = o0 - O0; }
  else                   { outBase = (size_t)NROWS * (O0 + O1); width = O2; oc0 = o0 - O0 - O1; }

#pragma unroll
  for (int tm = 0; tm < 2; ++tm) {
#pragma unroll
    for (int tn = 0; tn < 4; ++tn) {
      int r0 = m0 + wm * 32 + tm * 16 + lh * 8;
      int c  = oc0 + wn * 64 + tn * 16 + l15;
      float* op = out + outBase + (size_t)r0 * width + c;
#pragma unroll
      for (int j = 0; j < 8; ++j)
        op[(size_t)j * width] = acc[tm][tn][j];
    }
  }
}

// ---------------------------------------------------------------------------
// d_in order: 0:x 1:grid 2:base_w0 3:spline_w0 4:base_w1 5:spline_w1
//             6:base_w2 7:spline_w2.  Output: fp32, y0|y1|y2 concat.
// ---------------------------------------------------------------------------
extern "C" void kernel_launch(void* const* d_in, const int* in_sizes, int n_in,
                              void* d_out, int out_size, void* d_ws, size_t ws_size,
                              hipStream_t stream) {
  const float* x    = (const float*)d_in[0];
  const float* grid = (const float*)d_in[1];
  const float* bw0  = (const float*)d_in[2];
  const float* sw0  = (const float*)d_in[3];
  const float* bw1  = (const float*)d_in[4];
  const float* sw1  = (const float*)d_in[5];
  const float* bw2  = (const float*)d_in[6];
  const float* sw2  = (const float*)d_in[7];
  float* out = (float*)d_out;

  bf16_t* Ap = (bf16_t*)d_ws;                       // 8192*6912 bf16
  bf16_t* Wp = Ap + (size_t)NROWS * KTOT;           // 3072*6912 bf16

  {
    int total = NROWS * IN;
    kan_activations<<<(total + 255) / 256, 256, 0, stream>>>(x, grid, Ap);
  }
  {
    long total = (long)OTOT * KTOT;
    kan_pack_weights<<<(unsigned)((total + 255) / 256), 256, 0, stream>>>(
        bw0, sw0, bw1, sw1, bw2, sw2, Wp);
  }
  {
    dim3 g(OTOT / BN, NROWS / BM);                  // (24, 64)
    kan_gemm<<<g, 256, 0, stream>>>(Ap, Wp, out);
  }
}
